// UFOAttention_46497315946808
// MI455X (gfx1250) — matrix operand order repaired and checked by reading, not compiled, not run on hardware
//
#include <hip/hip_runtime.h>
#include <hip/hip_bf16.h>

// ---------------------------------------------------------------------------
// UFO linear attention for MI455X (gfx1250, wave32).
// All GEMMs run on v_wmma_f32_16x16x32_bf16 (fp32 accumulate), fed from
// LDS-staged tiles.  Tiles are filled with GLOBAL_LOAD_ASYNC_TO_LDS (ASYNCcnt)
// when the toolchain exposes the builtin, else with reg->ds_store copies.
// All B fragments of a chunk are materialized before the WMMA burst so the
// ds_loads overlap and the WMMAs issue back-to-back without hazard NOPs.
// ---------------------------------------------------------------------------

typedef __attribute__((ext_vector_type(16))) __bf16 v16bf;
typedef __attribute__((ext_vector_type(8)))  float  v8f;
typedef __attribute__((ext_vector_type(4)))  int    v4i;

#define DMODEL 1024
#define NHEAD  16
#define DK     64
#define DV     64
#define BATCH  4
#define SEQ    4096

#define AS1 __attribute__((address_space(1)))
#define AS3 __attribute__((address_space(3)))

#if defined(__has_builtin)
#if __has_builtin(__builtin_amdgcn_global_load_async_to_lds_b128)
#define HAVE_ASYNC_LDS 1
#endif
#endif
#ifndef HAVE_ASYNC_LDS
#define HAVE_ASYNC_LDS 0
#endif

static __device__ __forceinline__ v8f wmma_bf16(v16bf a, v16bf b, v8f c) {
    // (neg_a, A, neg_b, B, c_mod, C, reuse_a, reuse_b)
    return __builtin_amdgcn_wmma_f32_16x16x32_bf16(false, a, false, b, (short)0, c,
                                                   false, false);
}

// Copy 16 bytes global -> LDS. Async (ASYNCcnt) when available.
static __device__ __forceinline__ void stage16(float* ldst, const float* gsrc) {
#if HAVE_ASYNC_LDS
    __builtin_amdgcn_global_load_async_to_lds_b128((AS1 v4i*)gsrc, (AS3 v4i*)ldst,
                                                   0, 0);
#else
    *(float4*)ldst = *(const float4*)gsrc;
#endif
}

// Wait for this wave's outstanding stage16 transfers, then block barrier.
static __device__ __forceinline__ void stage_commit() {
#if HAVE_ASYNC_LDS
#if __has_builtin(__builtin_amdgcn_s_wait_asynccnt)
    __builtin_amdgcn_s_wait_asynccnt(0);
#else
    asm volatile("s_wait_asynccnt 0x0" ::: "memory");
#endif
#endif
    __syncthreads();
}

// A fragment (16x32 bf16) from a row-major f32 source with contiguous K.
// rowp = &A[m][kb] for this lane's row m = lane%16.
// lanes 0-15: elems 0..15 -> K+0..15; lanes 16-31: K+16..31.
static __device__ __forceinline__ v16bf frag_a_rowmajor(const float* rowp, int half) {
    const float* p0 = rowp + half * 16;
    const float* p1 = rowp + half * 16 + 8;
    v16bf a;
#pragma unroll
    for (int e = 0; e < 8; ++e) a[e] = (__bf16)p0[e];
#pragma unroll
    for (int e = 0; e < 8; ++e) a[8 + e] = (__bf16)p1[e];
    return a;
}

// B fragment (32x16 bf16) where column n is contiguous-K (NT GEMM from LDS).
// colp = &tile[n][0] (+kb); lanes 0-15 hold K 0..15, lanes 16-31 hold K 16..31.
static __device__ __forceinline__ v16bf frag_b_contig(const float* colp, int half) {
    const float* p = colp + half * 16;
    v16bf b;
#pragma unroll
    for (int e = 0; e < 16; ++e) b[e] = (__bf16)p[e];
    return b;
}

// ---------------------------------------------------------------------------
// C[M,N] = A[M,K] * W[N,K]^T + bias[N]
// 256 threads / 8 waves.  Block tile 128(M) x 64(N), BK = 32.
// A and W chunks double-buffered in LDS (f32, padded stride 36 -> conflict-free
// fragment reads), staged with async global->LDS copies.
// ---------------------------------------------------------------------------
#define GEMM_LDA 36
__global__ __launch_bounds__(256) void gemm_nt_bias(const float* __restrict__ A,
                                                    const float* __restrict__ W,
                                                    const float* __restrict__ bias,
                                                    float* __restrict__ C,
                                                    int M, int N, int K) {
    __shared__ __align__(16) float At[2][128 * GEMM_LDA];   // 36.9 KB
    __shared__ __align__(16) float Wt[2][64 * GEMM_LDA];    // 18.4 KB

    const int tid   = threadIdx.x;
    const int lane  = tid & 31;
    const int wave  = tid >> 5;
    const int half  = lane >> 4;
    const int l16   = lane & 15;
    const int mBase = blockIdx.x * 128;
    const int nBase = blockIdx.y * 64;

    // Cooperative staging map: one float4 per (row, quad) slot.
    const int srow = tid >> 3;            // 0..31
    const int sq   = (tid & 7) * 4;       // 0,4,..,28

    const int NC = K >> 5;                // 32-wide K chunks

    // Prologue: chunk 0 -> buffer 0.
#pragma unroll
    for (int it = 0; it < 4; ++it) {
        const int row = srow + it * 32;
        stage16(&At[0][row * GEMM_LDA + sq], A + (size_t)(mBase + row) * K + sq);
    }
#pragma unroll
    for (int it = 0; it < 2; ++it) {
        const int row = srow + it * 32;
        stage16(&Wt[0][row * GEMM_LDA + sq], W + (size_t)(nBase + row) * K + sq);
    }
    stage_commit();

    v8f acc[4] = {};
    for (int c = 0; c < NC; ++c) {
        const int buf = c & 1;
        const int nxt = buf ^ 1;
        if (c + 1 < NC) {
            const int kb = (c + 1) << 5;
#pragma unroll
            for (int it = 0; it < 4; ++it) {
                const int row = srow + it * 32;
                stage16(&At[nxt][row * GEMM_LDA + sq],
                        A + (size_t)(mBase + row) * K + kb + sq);
            }
#pragma unroll
            for (int it = 0; it < 2; ++it) {
                const int row = srow + it * 32;
                stage16(&Wt[nxt][row * GEMM_LDA + sq],
                        W + (size_t)(nBase + row) * K + kb + sq);
            }
        }
        // Materialize every fragment of this chunk, then burst the WMMAs.
        v16bf afrag = frag_a_rowmajor(&At[buf][(wave * 16 + l16) * GEMM_LDA], half);
        v16bf bfrag[4];
#pragma unroll
        for (int j = 0; j < 4; ++j)
            bfrag[j] = frag_b_contig(&Wt[buf][(j * 16 + l16) * GEMM_LDA], half);
#pragma unroll
        for (int j = 0; j < 4; ++j)
            acc[j] = wmma_bf16(afrag, bfrag[j], acc[j]);
        stage_commit();
    }

    const int mTile = mBase + wave * 16;
#pragma unroll
    for (int j = 0; j < 4; ++j) {
        const int n  = nBase + j * 16 + l16;
        const float bv = bias[n];
#pragma unroll
        for (int i = 0; i < 8; ++i) {
            const int row = mTile + half * 8 + i;
            C[(size_t)row * N + n] = acc[j][i] + bv;
        }
    }
}

// ---------------------------------------------------------------------------
// kv[b,h,dk,dv] = sum_n K[b,n,h,dk] * V[b,n,h,dv].  One block per (b,h).
// K/V chunks of 32 seq rows staged in LDS (shared by all 8 waves); every wave
// owns 2 of the 16 output 16x16 tiles over the full reduction.
// ---------------------------------------------------------------------------
__global__ __launch_bounds__(256) void kv_outer_kernel(const float* __restrict__ Kp,
                                                       const float* __restrict__ Vp,
                                                       float* __restrict__ KV) {
    __shared__ __align__(16) float Kt[2][32 * 64];   // 16 KB
    __shared__ __align__(16) float Vt[2][32 * 64];   // 16 KB

    const int tid   = threadIdx.x;
    const int lane  = tid & 31;
    const int wave  = tid >> 5;
    const int half  = lane >> 4;
    const int l16   = lane & 15;
    const int mtile = wave & 3;          // dk tile
    const int jpair = wave >> 2;         // dv tile pair
    const int bh    = blockIdx.x;        // 0..63
    const int b     = bh >> 4;
    const int h     = bh & 15;

    const size_t baseK = (size_t)b * SEQ * DMODEL + (size_t)h * DK;
    const size_t baseV = (size_t)b * SEQ * DMODEL + (size_t)h * DV;

    const int srow = tid >> 4;           // 0..15
    const int sq   = (tid & 15) * 4;     // 0..60

    const int NC = SEQ / 32;             // 128 chunks

#pragma unroll
    for (int it = 0; it < 2; ++it) {
        const int row = srow + it * 16;
        stage16(&Kt[0][row * 64 + sq], Kp + baseK + (size_t)row * DMODEL + sq);
        stage16(&Vt[0][row * 64 + sq], Vp + baseV + (size_t)row * DMODEL + sq);
    }
    stage_commit();

    const int m = mtile * 16 + l16;      // dk row for A fragment
    v8f acc[2] = {};
    for (int c = 0; c < NC; ++c) {
        const int buf = c & 1;
        const int nxt = buf ^ 1;
        if (c + 1 < NC) {
            const size_t n0 = (size_t)(c + 1) * 32;
#pragma unroll
            for (int it = 0; it < 2; ++it) {
                const int row = srow + it * 16;
                stage16(&Kt[nxt][row * 64 + sq],
                        Kp + baseK + (n0 + row) * DMODEL + sq);
                stage16(&Vt[nxt][row * 64 + sq],
                        Vp + baseV + (n0 + row) * DMODEL + sq);
            }
        }
        // A = K^T : A[m][n] = Kt[n][m]
        const float* kt = Kt[buf];
        v16bf a;
#pragma unroll
        for (int e = 0; e < 8; ++e) a[e] = (__bf16)kt[(half * 16 + e) * 64 + m];
#pragma unroll
        for (int e = 0; e < 8; ++e) a[8 + e] = (__bf16)kt[(half * 16 + 8 + e) * 64 + m];

        const float* vt = Vt[buf];
        v16bf bfr[2];
#pragma unroll
        for (int jj = 0; jj < 2; ++jj) {
            const int col = (jpair * 2 + jj) * 16 + l16;
#pragma unroll
            for (int e = 0; e < 16; ++e)
                bfr[jj][e] = (__bf16)vt[(half * 16 + e) * 64 + col];
        }
#pragma unroll
        for (int jj = 0; jj < 2; ++jj)
            acc[jj] = wmma_bf16(a, bfr[jj], acc[jj]);
        stage_commit();
    }

    float* out = KV + (size_t)bh * (DK * DV);
#pragma unroll
    for (int jj = 0; jj < 2; ++jj) {
        const int col = (jpair * 2 + jj) * 16 + l16;
#pragma unroll
        for (int i = 0; i < 8; ++i) {
            const int row = mtile * 16 + half * 8 + i;
            out[row * 64 + col] = acc[jj][i];
        }
    }
}

// ---------------------------------------------------------------------------
// In-place xnorm over rows of length 64: x = x * gamma[h] / ||x||.
// One wave per row; h = (row / h_div) % 16.
// ---------------------------------------------------------------------------
__global__ __launch_bounds__(256) void xnorm64_kernel(float* __restrict__ X,
                                                      const float* __restrict__ gamma,
                                                      int nrows, int h_div) {
    const int lane = threadIdx.x & 31;
    const int wave = threadIdx.x >> 5;
    const int r    = blockIdx.x * 8 + wave;
    if (r >= nrows) return;
    float* row = X + (size_t)r * 64;
    float x0 = row[lane];
    float x1 = row[lane + 32];
    float s  = x0 * x0 + x1 * x1;
#pragma unroll
    for (int off = 16; off; off >>= 1) s += __shfl_xor(s, off, 32);
    const int h = (r / h_div) & (NHEAD - 1);
    const float scale = gamma[h] / sqrtf(s);
    row[lane]      = x0 * scale;
    row[lane + 32] = x1 * scale;
}

// ---------------------------------------------------------------------------
// out[b,n,h*64+dv] = sum_dk qn[b,n,h,dk] * kvn[b,h,dk,dv]
// Grid (SEQ/128, H, B).  kv_n (16 KB) staged once in LDS per block; q rows
// stream from global as contiguous-K A fragments.
// ---------------------------------------------------------------------------
__global__ __launch_bounds__(256) void attn_apply_kernel(const float* __restrict__ Qn,
                                                         const float* __restrict__ KVn,
                                                         float* __restrict__ Out) {
    __shared__ __align__(16) float KVs[64 * 64];     // 16 KB

    const int tid  = threadIdx.x;
    const int lane = tid & 31;
    const int wave = tid >> 5;
    const int half = lane >> 4;
    const int l16  = lane & 15;
    const int h = blockIdx.y;
    const int b = blockIdx.z;

    const float* Bm = KVn + (size_t)(b * NHEAD + h) * (DK * DV);
    const int srow = tid >> 4;           // 0..15
    const int sq   = (tid & 15) * 4;
#pragma unroll
    for (int it = 0; it < 4; ++it) {
        const int row = srow + it * 16;
        stage16(&KVs[row * 64 + sq], Bm + (size_t)row * 64 + sq);
    }
    stage_commit();

    const int rowTile = blockIdx.x * 128 + wave * 16;
    const float* arow = Qn + (size_t)b * SEQ * DMODEL +
                        (size_t)(rowTile + l16) * DMODEL + h * DK;

    v8f acc[4] = {};
#pragma unroll
    for (int kb = 0; kb < DK; kb += 32) {
        v16bf a = frag_a_rowmajor(arow + kb, half);
        v16bf bfr[4];
#pragma unroll
        for (int j = 0; j < 4; ++j) {
            const int col = j * 16 + l16;
#pragma unroll
            for (int e = 0; e < 16; ++e)
                bfr[j][e] = (__bf16)KVs[(kb + half * 16 + e) * 64 + col];
        }
#pragma unroll
        for (int j = 0; j < 4; ++j)
            acc[j] = wmma_bf16(a, bfr[j], acc[j]);
    }
    float* outBase = Out + (size_t)b * SEQ * DMODEL + (size_t)h * DV;
#pragma unroll
    for (int j = 0; j < 4; ++j) {
        const int n = j * 16 + l16;
#pragma unroll
        for (int i = 0; i < 8; ++i) {
            const int row = rowTile + half * 8 + i;
            outBase[(size_t)row * DMODEL + n] = acc[j][i];
        }
    }
}

// ---------------------------------------------------------------------------
extern "C" void kernel_launch(void* const* d_in, const int* in_sizes, int n_in,
                              void* d_out, int out_size, void* d_ws, size_t ws_size,
                              hipStream_t stream) {
    (void)in_sizes; (void)n_in; (void)out_size; (void)ws_size;
    const float* queries = (const float*)d_in[0];
    const float* keys    = (const float*)d_in[1];
    const float* values  = (const float*)d_in[2];
    const float* Wq = (const float*)d_in[3];
    const float* bq = (const float*)d_in[4];
    const float* Wk = (const float*)d_in[5];
    const float* bk = (const float*)d_in[6];
    const float* Wv = (const float*)d_in[7];
    const float* bv = (const float*)d_in[8];
    const float* Wo = (const float*)d_in[9];
    const float* bo = (const float*)d_in[10];
    const float* gamma = (const float*)d_in[11];

    const int M = BATCH * SEQ;                 // 16384 flattened (b,n)
    const size_t MN = (size_t)M * DMODEL;

    float* ws  = (float*)d_ws;
    float* q   = ws;                           // [B,N,H,DK] -> normalized in place
    float* k   = ws + MN;                      // [B,N,H,DK]
    float* v   = ws + 2 * MN;                  // [B,N,H,DV] -> reused for attn out
    float* kv  = ws + 3 * MN;                  // [B,H,DK,DV], normalized in place
    float* attn = v;                           // v is dead once kv is built

    dim3 blk(256);
    dim3 gGemm(M / 128, DMODEL / 64);

    // 1-3. Q/K/V projections.
    hipLaunchKernelGGL(gemm_nt_bias, gGemm, blk, 0, stream, queries, Wq, bq, q, M, DMODEL, DMODEL);
    hipLaunchKernelGGL(gemm_nt_bias, gGemm, blk, 0, stream, keys,    Wk, bk, k, M, DMODEL, DMODEL);
    hipLaunchKernelGGL(gemm_nt_bias, gGemm, blk, 0, stream, values,  Wv, bv, v, M, DMODEL, DMODEL);

    // 4. kv = K^T V per (b,h).
    hipLaunchKernelGGL(kv_outer_kernel, dim3(BATCH * NHEAD), blk, 0, stream, k, v, kv);

    // 5. xnorm(kv): rows = B*H*DK = 4096, h = (row/64)%16.
    hipLaunchKernelGGL(xnorm64_kernel, dim3(BATCH * NHEAD * DK / 8), blk, 0, stream,
                       kv, gamma, BATCH * NHEAD * DK, DK);

    // 6. xnorm(q): rows = B*N*H = 262144, h = row%16.
    hipLaunchKernelGGL(xnorm64_kernel, dim3(M * NHEAD / 8), blk, 0, stream,
                       q, gamma, M * NHEAD, 1);

    // 7. out = q_n @ kv_n per (b,h).
    hipLaunchKernelGGL(attn_apply_kernel, dim3(SEQ / 128, NHEAD, BATCH), blk, 0, stream,
                       q, kv, attn);

    // 8. Output projection into d_out.
    hipLaunchKernelGGL(gemm_nt_bias, gGemm, blk, 0, stream,
                       attn, Wo, bo, (float*)d_out, M, DMODEL, DMODEL);
}
